// TiSasRecLayers_38585986187988
// MI455X (gfx1250) — matrix operand order repaired and checked by reading, not compiled, not run on hardware
//
#include <hip/hip_runtime.h>
#include <hip/hip_bf16.h>

typedef __attribute__((ext_vector_type(16))) _Float16 v16h;
typedef __attribute__((ext_vector_type(8)))  float    v8f;

#define BB 32
#define LL 200
#define HH 64
#define NHH 2
#define DD 32
#define BL (BB*LL)          // 6400 rows
#define NKT 13              // ceil(L/16)
#define KPAD 224            // ceil(L/32)*32

// Pack two contiguous 8-float runs into one f16 A/B fragment (WMMA 16x16x32).
// Element pattern: frag[0..7] = p0[0..7], frag[8..15] = p1[0..7].
__device__ __forceinline__ v16h pack16(const float* __restrict__ p0,
                                       const float* __restrict__ p1) {
  float4 x0 = *(const float4*)(p0);
  float4 x1 = *(const float4*)(p0 + 4);
  float4 y0 = *(const float4*)(p1);
  float4 y1 = *(const float4*)(p1 + 4);
  v16h a;
  a[0] = (_Float16)x0.x;  a[1] = (_Float16)x0.y;
  a[2] = (_Float16)x0.z;  a[3] = (_Float16)x0.w;
  a[4] = (_Float16)x1.x;  a[5] = (_Float16)x1.y;
  a[6] = (_Float16)x1.z;  a[7] = (_Float16)x1.w;
  a[8] = (_Float16)y0.x;  a[9] = (_Float16)y0.y;
  a[10] = (_Float16)y0.z; a[11] = (_Float16)y0.w;
  a[12] = (_Float16)y1.x; a[13] = (_Float16)y1.y;
  a[14] = (_Float16)y1.z; a[15] = (_Float16)y1.w;
  return a;
}

// ---------------------------------------------------------------------------
// LayerNorm (compile-time optional residual add): one wave per 64-wide row.
// ---------------------------------------------------------------------------
template <bool WITH_ADD>
__global__ __launch_bounds__(256) void ln_kernel(const float* __restrict__ x,
                                                 const float* __restrict__ add,
                                                 const float* __restrict__ g,
                                                 const float* __restrict__ bb,
                                                 float* __restrict__ out) {
  int wave = threadIdx.x >> 5, lane = threadIdx.x & 31;
  int row  = blockIdx.x * 8 + wave;
  const float* xr = x + (size_t)row * HH;
  float e0 = xr[lane], e1 = xr[lane + 32];
  if (WITH_ADD) {
    e0 += add[(size_t)row * HH + lane];
    e1 += add[(size_t)row * HH + lane + 32];
  }
  float s = e0 + e1;
  for (int m = 16; m; m >>= 1) s += __shfl_xor(s, m, 32);
  float mean = s * (1.0f / 64.0f);
  float d0 = e0 - mean, d1 = e1 - mean;
  float v = d0 * d0 + d1 * d1;
  for (int m = 16; m; m >>= 1) v += __shfl_xor(v, m, 32);
  float rstd = rsqrtf(v * (1.0f / 64.0f) + 1e-8f);
  out[(size_t)row * HH + lane]      = d0 * rstd * g[lane]      + bb[lane];
  out[(size_t)row * HH + lane + 32] = d1 * rstd * g[lane + 32] + bb[lane + 32];
}

// ---------------------------------------------------------------------------
// WMMA GEMM: out[BL,64] = A[BL,64] @ W[64,64] + bias (+extra)(relu)(+resid)(*pad)
// One 16x16 C tile per wave, K=64 => two v_wmma_f32_16x16x32_f16.
// ---------------------------------------------------------------------------
template <bool EXTRA, bool RELU, bool RESID, bool PADM>
__global__ __launch_bounds__(256) void gemm64_kernel(const float* __restrict__ A,
                                                     const float* __restrict__ W,
                                                     const float* __restrict__ bias,
                                                     const float* __restrict__ extra,
                                                     const float* __restrict__ resid,
                                                     const unsigned char* __restrict__ pad,
                                                     float* __restrict__ out) {
  __shared__ float Ws[HH * HH];
  __shared__ float bs[HH];
  int tid = threadIdx.x;
  for (int i = tid; i < HH * HH; i += 256) Ws[i] = W[i];
  if (tid < HH) bs[tid] = bias[tid];
  __syncthreads();

  int wave = tid >> 5, lane = tid & 31;
  int mt = blockIdx.x * 2 + (wave >> 2);   // 2 M-tiles per block
  int nt = wave & 3;                       // 4 N-tiles cover 64 cols
  int m0 = mt * 16, n0 = nt * 16;
  int ml = lane & 15, hi = lane >> 4;

  const float* Arow = A + (size_t)(m0 + ml) * HH + hi * 8;
  v16h a0 = pack16(Arow,      Arow + 16);
  v16h a1 = pack16(Arow + 32, Arow + 48);

  int n = n0 + ml;
  v16h b0, b1;
#pragma unroll
  for (int i = 0; i < 8; ++i) {
    int k = 2 * i + hi * 8 + (i >> 2) * 8;
    b0[2 * i]     = (_Float16)Ws[k * HH + n];
    b0[2 * i + 1] = (_Float16)Ws[(k + 1) * HH + n];
    b1[2 * i]     = (_Float16)Ws[(32 + k) * HH + n];
    b1[2 * i + 1] = (_Float16)Ws[(33 + k) * HH + n];
  }
  v8f c = {};
  c = __builtin_amdgcn_wmma_f32_16x16x32_f16(false, a0, false, b0, (short)0, c, false, false);
  c = __builtin_amdgcn_wmma_f32_16x16x32_f16(false, a1, false, b1, (short)0, c, false, false);

  // Branch-free epilogue: gather optional terms first (loads pipeline), then store.
  float ext[8], res[8], padv[8];
#pragma unroll
  for (int r = 0; r < 8; ++r) {
    int row = m0 + r + hi * 8;
    if (EXTRA) ext[r] = extra[(size_t)row * HH + n];
    if (RESID) res[r] = resid[(size_t)row * HH + n];
    if (PADM)  padv[r] = pad[row] ? 1.0f : 0.0f;
  }
#pragma unroll
  for (int r = 0; r < 8; ++r) {
    int row = m0 + r + hi * 8;
    float v = c[r] + bs[n];
    if (EXTRA) v += ext[r];
    if (RELU)  v = v > 0.0f ? v : 0.0f;
    if (RESID) v += res[r];
    if (PADM)  v *= padv[r];
    out[(size_t)row * HH + n] = v;
  }
}

// ---------------------------------------------------------------------------
// Fused attention: scores (WMMA QK^T + streamed tmk dot) -> softmax ->
// output (WMMA P*Veff + streamed tmv weighted sum). Causal-clipped streaming.
// grid = (13 q-tiles, B, NH), block = 256 (8 waves).
// ---------------------------------------------------------------------------
__global__ __launch_bounds__(256) void attn_kernel(const float* __restrict__ qh,
                                                   const float* __restrict__ keff,
                                                   const float* __restrict__ veff,
                                                   const float* __restrict__ tmk,
                                                   const float* __restrict__ tmv,
                                                   const unsigned char* __restrict__ pad,
                                                   float* __restrict__ att) {
  const int h = blockIdx.z, b = blockIdx.y, q0 = blockIdx.x * 16;
  __shared__ float S[16][KPAD];
  __shared__ float Qs[16][DD];
  __shared__ float Obuf[16][DD];
  int tid = threadIdx.x, wave = tid >> 5, lane = tid & 31;
  int ml = lane & 15, hi = lane >> 4;

  // stage Q tile
  for (int i = tid; i < 16 * DD; i += 256) {
    int q = i >> 5, d = i & 31, qg = q0 + q;
    Qs[q][d] = (qg < LL) ? qh[(size_t)(b * LL + qg) * HH + h * DD + d] : 0.0f;
  }
  __syncthreads();

  // ---- phase 1a: WMMA scores S = Q * Keff^T (skip fully-masked k-tiles).
  // Out-of-range keys are CLAMPED, not zeroed: every S column is rewritten by
  // the masking pass in phase 2, so clamped-key scores never survive.
  int qmax = (q0 + 15 < LL) ? (q0 + 15) : (LL - 1);
  int lastkt = qmax >> 4;
  for (int kt = wave; kt <= lastkt; kt += 8) {
    v16h a = pack16(&Qs[ml][hi * 16], &Qs[ml][hi * 16 + 8]);
    int kk = kt * 16 + ml;
    int kcl = kk < LL ? kk : LL - 1;
    const float* kr = keff + (size_t)(b * LL + kcl) * HH + h * DD + hi * 16;
    v16h bf = pack16(kr, kr + 8);
    v8f c = {};
    c = __builtin_amdgcn_wmma_f32_16x16x32_f16(false, a, false, bf, (short)0, c, false, false);
#pragma unroll
    for (int r = 0; r < 8; ++r) S[r + hi * 8][kt * 16 + ml] = c[r];
  }
  // zero padding columns beyond the last written k-tile (208..223 and any
  // tiles skipped by the causal clip) so phase 2 reads defined values.
  __syncthreads();
  for (int i = tid; i < 16 * (KPAD - 208) ; i += 256) {
    int q = i / (KPAD - 208), k = 208 + i % (KPAD - 208);
    S[q][k] = 0.0f;
  }
  for (int kt = lastkt + 1; kt < NKT; ++kt)
    for (int i = tid; i < 16 * 16; i += 256)
      S[i >> 4][kt * 16 + (i & 15)] = 0.0f;
  __syncthreads();

  // ---- phase 1b: S[q,k] += dot(tmk[b,q,k,h*32:], Q[q,:]) for k <= q ----
  for (int j = 0; j < 2; ++j) {
    int lq = wave * 2 + j, qg = q0 + lq;
    if (qg < LL) {
      const float* base = tmk + (size_t)(b * LL + qg) * LL * HH + h * DD;
      for (int k = lane; k <= qg; k += 32) {
        const float* t = base + (size_t)k * HH;
        __builtin_prefetch(base + (size_t)(k + 64) * HH, 0, 1);
        float dot = 0.0f;
#pragma unroll
        for (int d = 0; d < DD; d += 4) {
          float4 tv = *(const float4*)(t + d);
          dot += tv.x * Qs[lq][d] + tv.y * Qs[lq][d + 1] +
                 tv.z * Qs[lq][d + 2] + tv.w * Qs[lq][d + 3];
        }
        S[lq][k] += dot;
      }
    }
  }
  __syncthreads();

  // ---- phase 2: per-row softmax with causal + padding masks ----
  const float scale = 0.17677669529663687f;   // 1/sqrt(D)
  const float NEGV = -4294967295.0f;          // -2^32 + 1
  for (int j = 0; j < 2; ++j) {
    int lq = wave * 2 + j, qg = q0 + lq;
    if (qg < LL) {
      bool pr = pad[b * LL + qg] != 0;
      float mx = NEGV;
      for (int k = lane; k < KPAD; k += 32) {
        float s = (pr && k <= qg) ? S[lq][k] * scale : NEGV;
        S[lq][k] = s;
        mx = fmaxf(mx, s);
      }
      for (int m = 16; m; m >>= 1) mx = fmaxf(mx, __shfl_xor(mx, m, 32));
      float sum = 0.0f;
      for (int k = lane; k < KPAD; k += 32) {
        float e = __expf(S[lq][k] - mx);
        S[lq][k] = e;
        sum += e;
      }
      for (int m = 16; m; m >>= 1) sum += __shfl_xor(sum, m, 32);
      float inv = 1.0f / sum;
      for (int k = lane; k < KPAD; k += 32) S[lq][k] *= inv;
    }
  }
  __syncthreads();

  // ---- phase 3a: WMMA O = P * Veff (waves 0,1 own the two 16-col d-tiles).
  // Rows with k >= L are clamped: their P entries are exactly 0 after softmax,
  // so clamped V rows contribute 0.
  if (wave < 2) {
    int n0 = wave * 16;
    int col = n0 + ml;
    v8f c = {};
    int kcmax = qmax >> 5;
    const float* vbase = veff + (size_t)(b * LL) * HH + h * DD + col;
    for (int kc = 0; kc <= kcmax; ++kc) {
      v16h a = pack16(&S[ml][kc * 32 + hi * 16], &S[ml][kc * 32 + hi * 16 + 8]);
      v16h bf;
#pragma unroll
      for (int i = 0; i < 16; ++i) {
        int k = kc * 32 + i + hi * 16;
        int kcl = k < LL ? k : LL - 1;
        bf[i] = (_Float16)vbase[(size_t)kcl * HH];
      }
      c = __builtin_amdgcn_wmma_f32_16x16x32_f16(false, a, false, bf, (short)0, c, false, false);
    }
#pragma unroll
    for (int r = 0; r < 8; ++r) Obuf[r + hi * 8][n0 + ml] = c[r];
  }

  // ---- phase 3b: streamed tmv weighted sum, lane = d (coalesced 128B) ----
  float acc[2] = {0.0f, 0.0f};
  for (int j = 0; j < 2; ++j) {
    int lq = wave * 2 + j, qg = q0 + lq;
    if (qg < LL) {
      const float* base = tmv + (size_t)(b * LL + qg) * LL * HH + h * DD + lane;
      float a = 0.0f;
      for (int k = 0; k <= qg; ++k) {
        __builtin_prefetch(base + (size_t)(k + 16) * HH, 0, 1);
        a += S[lq][k] * base[(size_t)k * HH];
      }
      acc[j] = a;
    }
  }
  __syncthreads();
  for (int j = 0; j < 2; ++j) {
    int lq = wave * 2 + j, qg = q0 + lq;
    if (qg < LL) Obuf[lq][lane] += acc[j];
  }
  __syncthreads();

  for (int i = tid; i < 16 * DD; i += 256) {
    int q = i >> 5, d = i & 31, qg = q0 + q;
    if (qg < LL) att[(size_t)(b * LL + qg) * HH + h * DD + d] = Obuf[q][d];
  }
}

// ---------------------------------------------------------------------------
extern "C" void kernel_launch(void* const* d_in, const int* in_sizes, int n_in,
                              void* d_out, int out_size, void* d_ws, size_t ws_size,
                              hipStream_t stream) {
  const float* seqs = (const float*)d_in[0];
  const unsigned char* pad = (const unsigned char*)d_in[2];   // bool [B,L,1]
  const float* tmk = (const float*)d_in[3];
  const float* tmv = (const float*)d_in[4];
  const float* pk  = (const float*)d_in[5];
  const float* pv  = (const float*)d_in[6];
  const float* Wq  = (const float*)d_in[7];
  const float* bq  = (const float*)d_in[8];
  const float* Wk  = (const float*)d_in[9];
  const float* bk  = (const float*)d_in[10];
  const float* Wv  = (const float*)d_in[11];
  const float* bv  = (const float*)d_in[12];
  const float* g1  = (const float*)d_in[13];
  const float* be1 = (const float*)d_in[14];
  const float* g2  = (const float*)d_in[15];
  const float* be2 = (const float*)d_in[16];
  const float* W1  = (const float*)d_in[17];
  const float* bb1 = (const float*)d_in[18];
  const float* W2  = (const float*)d_in[19];
  const float* bb2 = (const float*)d_in[20];

  float* ws = (float*)d_ws;
  const size_t N = (size_t)BL * HH;          // 409600 floats per buffer
  float* qbuf = ws + 0 * N;
  float* qhb  = ws + 1 * N;
  float* keff = ws + 2 * N;
  float* veff = ws + 3 * N;
  float* attb = ws + 4 * N;
  float* x2   = ws + 5 * N;
  float* h1   = ws + 6 * N;
  float* xbuf = ws + 7 * N;
  float* outp = (float*)d_out;

  const float* xin = seqs;
  for (int blk = 0; blk < 2; ++blk) {
    // q = LN1(x)
    ln_kernel<false><<<BL / 8, 256, 0, stream>>>(xin, nullptr, g1, be1, qbuf);
    // projections (pos vectors folded into K/V)
    gemm64_kernel<false, false, false, false>
        <<<BL / 32, 256, 0, stream>>>(qbuf, Wq, bq, nullptr, nullptr, nullptr, qhb);
    gemm64_kernel<true, false, false, false>
        <<<BL / 32, 256, 0, stream>>>(xin, Wk, bk, pk, nullptr, nullptr, keff);
    gemm64_kernel<true, false, false, false>
        <<<BL / 32, 256, 0, stream>>>(xin, Wv, bv, pv, nullptr, nullptr, veff);
    // fused attention (single pass over time matrices, causal-clipped)
    attn_kernel<<<dim3(NKT, BB, NHH), 256, 0, stream>>>(qhb, keff, veff, tmk, tmv, pad, attb);
    // x = LN2(q + att)
    ln_kernel<true><<<BL / 8, 256, 0, stream>>>(qbuf, attb, g2, be2, x2);
    // FFN + residual + padding mask
    gemm64_kernel<false, true, false, false>
        <<<BL / 32, 256, 0, stream>>>(x2, W1, bb1, nullptr, nullptr, nullptr, h1);
    float* xo = (blk == 1) ? outp : xbuf;
    gemm64_kernel<false, false, true, true>
        <<<BL / 32, 256, 0, stream>>>(h1, W2, bb2, nullptr, x2, pad, xo);
    xin = xbuf;
  }
}
